// Aggregator_63015760167158
// MI455X (gfx1250) — matrix operand order repaired and checked by reading, not compiled, not run on hardware
//
#include <hip/hip_runtime.h>

#define DIM 128

typedef __attribute__((ext_vector_type(16))) _Float16 v16h;
typedef __attribute__((ext_vector_type(8)))  float    v8f;

// ---------------------------------------------------------------------------
// Phase 0: zero the fp32 side-accumulator (ws is poisoned by the harness).
// ---------------------------------------------------------------------------
__global__ void zero_f4(float4* __restrict__ p, int n4) {
  int i = blockIdx.x * blockDim.x + threadIdx.x;
  if (i < n4) p[i] = make_float4(0.f, 0.f, 0.f, 0.f);
}

// ---------------------------------------------------------------------------
// Phase 1: COO SpMM  side[rows[e]] += vals[e] * ego[cols[e]]
// One wave32 per edge; each lane moves 4 contiguous floats (float4 gather,
// 4x global_atomic_add_f32 scatter). ego/side are L2-resident (25.6MB each).
// ---------------------------------------------------------------------------
__global__ __launch_bounds__(256) void spmm_coo(
    const float* __restrict__ ego, const float* __restrict__ vals,
    const int* __restrict__ rows, const int* __restrict__ cols,
    float* __restrict__ side, int n_edges) {
  int gtid = blockIdx.x * blockDim.x + threadIdx.x;
  int e    = gtid >> 5;
  int lane = threadIdx.x & 31;
  if (e >= n_edges) return;
  int   r = rows[e];
  int   c = cols[e];
  float v = vals[e];
  const float4* __restrict__ src = (const float4*)(ego + (size_t)c * DIM);
  float*        __restrict__ dst = side + (size_t)r * DIM + lane * 4;
  float4 x = src[lane];
  atomicAdd(dst + 0, v * x.x);
  atomicAdd(dst + 1, v * x.y);
  atomicAdd(dst + 2, v * x.z);
  atomicAdd(dst + 3, v * x.w);
}

// ---------------------------------------------------------------------------
// Phase 2: out = lrelu((ego+side)@W1^T + b1) + lrelu((ego*side)@W2^T + b2)
// f16 WMMA (v_wmma_f32_16x16x32_f16), fp32 accumulation. f16 inputs carry
// more mantissa than the reference's own bf16 round-trip of `side`, and the
// fp32->f16 convert is a single HW instruction (v_cvt_pk_f16_f32 / fma_mix).
// 8 waves/block; each wave owns one 16-row M-tile, full N=128.
// W1/W2 staged once per block as f16 in LDS (2 x 32KB; WGP LDS = 320KB).
// Zero-init accumulators keep the first WMMA's C as an inline constant 0;
// bias loads stay inside the nt-loop so no long-lived registers force the
// compiler to hoist/spill the loop-invariant LDS B-fragments (observed in a
// prior variant as scratch_load_b128 reloads).
// Requires n_nodes % 16 == 0 (50000 = 3125*16).
// ---------------------------------------------------------------------------
__global__ __launch_bounds__(256) void fused_bi_interaction(
    const float* __restrict__ ego, const float* __restrict__ side,
    const float* __restrict__ W1, const float* __restrict__ b1,
    const float* __restrict__ W2, const float* __restrict__ b2,
    float* __restrict__ out, int n_tiles) {
  __shared__ _Float16 sW1[DIM * DIM];   // 32 KB, row-major [n][k] (torch layout)
  __shared__ _Float16 sW2[DIM * DIM];   // 32 KB

  // Cooperative f32 -> f16 conversion of both weight matrices.
  for (int i = threadIdx.x * 4; i < DIM * DIM; i += blockDim.x * 4) {
    float4 w1 = *(const float4*)(W1 + i);
    float4 w2 = *(const float4*)(W2 + i);
    sW1[i + 0] = (_Float16)w1.x; sW1[i + 1] = (_Float16)w1.y;
    sW1[i + 2] = (_Float16)w1.z; sW1[i + 3] = (_Float16)w1.w;
    sW2[i + 0] = (_Float16)w2.x; sW2[i + 1] = (_Float16)w2.y;
    sW2[i + 2] = (_Float16)w2.z; sW2[i + 3] = (_Float16)w2.w;
  }
  __syncthreads();

  const int wave = threadIdx.x >> 5;   // 0..7 (wave-uniform)
  const int lane = threadIdx.x & 31;
  const int half = lane >> 4;          // selects K-half per the A/B layouts
  const int l16  = lane & 15;         // M (for A) / N (for B,C,D)

  for (int t = blockIdx.x * 8 + wave; t < n_tiles; t += gridDim.x * 8) {
    // ---- Build A fragments for both GEMMs (contiguous K order, same as B):
    // lane(0..15)=M, element j holds K = ks*32 + half*16 + j.
    const float4* erow = (const float4*)(ego  + (size_t)(t * 16 + l16) * DIM);
    const float4* srow = (const float4*)(side + (size_t)(t * 16 + l16) * DIM);
    v16h a1[4], a2[4];
#pragma unroll
    for (int ks = 0; ks < 4; ++ks) {
      int q = ks * 8 + half * 4;       // float4 index of K = ks*32 + half*16
      float ef[16], sf[16];
      *(float4*)&ef[0]  = erow[q];     *(float4*)&ef[4]  = erow[q + 1];
      *(float4*)&ef[8]  = erow[q + 2]; *(float4*)&ef[12] = erow[q + 3];
      *(float4*)&sf[0]  = srow[q];     *(float4*)&sf[4]  = srow[q + 1];
      *(float4*)&sf[8]  = srow[q + 2]; *(float4*)&sf[12] = srow[q + 3];
#pragma unroll
      for (int j = 0; j < 16; ++j) {
        a1[ks][j] = (_Float16)(ef[j] + sf[j]);   // sum-branch input
        a2[ks][j] = (_Float16)(ef[j] * sf[j]);   // bi-branch input
      }
    }

    // ---- Sweep 8 N-tiles; per tile 4 K-steps x 2 GEMMs = 8 WMMAs.
#pragma unroll
    for (int nt = 0; nt < 8; ++nt) {
      const int n = nt * 16 + l16;     // output column for this lane
      v8f acc1 = {}, acc2 = {};
#pragma unroll
      for (int ks = 0; ks < 4; ++ks) {
        // B fragment, contiguous K order as in A: lane N=l16, half selects K 0..15 vs 16..31.
        // B[k][n] = W[n][k] -> contiguous 16 halfs from LDS row n.
        int koff = ks * 32 + half * 16;
        v16h bf1 = *(const v16h*)(sW1 + n * DIM + koff);
        v16h bf2 = *(const v16h*)(sW2 + n * DIM + koff);
        acc1 = __builtin_amdgcn_wmma_f32_16x16x32_f16(
            false, a1[ks], false, bf1, (short)0, acc1, false, false);
        acc2 = __builtin_amdgcn_wmma_f32_16x16x32_f16(
            false, a2[ks], false, bf2, (short)0, acc2, false, false);
      }
      // ---- Epilogue: bias + leaky_relu(0.01) + sum.
      // C/D layout: VGPR r -> row M = half*8 + r, col N = l16.
      float bias1 = b1[n];
      float bias2 = b2[n];
      float* op = out + (size_t)(t * 16 + half * 8) * DIM + n;
#pragma unroll
      for (int r = 0; r < 8; ++r) {
        float x = acc1[r] + bias1; x = (x > 0.f) ? x : 0.01f * x;
        float y = acc2[r] + bias2; y = (y > 0.f) ? y : 0.01f * y;
        op[(size_t)r * DIM] = x + y;
      }
    }
  }
}

extern "C" void kernel_launch(void* const* d_in, const int* in_sizes, int n_in,
                              void* d_out, int out_size, void* d_ws, size_t ws_size,
                              hipStream_t stream) {
  const float* ego  = (const float*)d_in[0];
  const float* vals = (const float*)d_in[1];
  const float* W1   = (const float*)d_in[2];
  const float* b1   = (const float*)d_in[3];
  const float* W2   = (const float*)d_in[4];
  const float* b2   = (const float*)d_in[5];
  const int*   rows = (const int*)d_in[6];
  const int*   cols = (const int*)d_in[7];
  float*       out  = (float*)d_out;
  float*       side = (float*)d_ws;          // [n_nodes, DIM] fp32 accumulator

  const int n_nodes = in_sizes[0] / DIM;     // 50000
  const int n_edges = in_sizes[1];           // 640000

  // Phase 0: zero side accumulator.
  const int n4 = n_nodes * DIM / 4;
  zero_f4<<<(n4 + 255) / 256, 256, 0, stream>>>((float4*)side, n4);

  // Phase 1: COO SpMM with L2-resident float atomics (8 edges per 256-thr block).
  const int spmm_blocks = (n_edges + 7) / 8;
  spmm_coo<<<spmm_blocks, 256, 0, stream>>>(ego, vals, rows, cols, side, n_edges);

  // Phase 2: fused dual-GEMM + bias + leaky_relu + add via f16 WMMA.
  const int n_tiles = n_nodes / 16;          // 3125 (exact)
  const int gemm_blocks = (n_tiles + 7) / 8; // 8 waves/block, 1 tile/wave
  fused_bi_interaction<<<gemm_blocks, 256, 0, stream>>>(
      ego, side, W1, b1, W2, b2, out, n_tiles);
}
